// Generator_59382217835188
// MI455X (gfx1250) — matrix operand order repaired and checked by reading, not compiled, not run on hardware
//
#include <hip/hip_runtime.h>
#include <hip/hip_bf16.h>
#include <cstddef>

// ---------------------------------------------------------------------------
// MI455X (gfx1250, wave32) implementation.
// Heavy GEMMs use v_wmma_f32_16x16x32_bf16 (f32 accumulate) with LDS-staged
// 64x64 block tiles holding PACKED bf16 pairs (conversion done once at fill
// time -> inner loop is pure ds_load + wmma; guard-free fast fill for
// interior tiles). Sequential GRU scans are latency-bound persistent blocks.
// MAC loop is algebraically reduced to matvecs with loop-invariant GEMMs
// hoisted.
// ---------------------------------------------------------------------------

#define H_   300
#define D_   600
#define VOC  19495
#define LKB  2048
#define LQ   512
#define LT   2048
#define NMAC 7

typedef __attribute__((ext_vector_type(16))) __bf16 v16bf;
typedef __attribute__((ext_vector_type(8)))  float  v8f;

// pack two fp32 into adjacent bf16 halves (RNE), lo -> [15:0], hi -> [31:16]
static __device__ inline unsigned pack2(float lo, float hi) {
  union { float f; unsigned u; } a, b;
  a.f = lo; b.f = hi;
  unsigned ul = a.u + 0x7FFFu + ((a.u >> 16) & 1u);
  unsigned uh = b.u + 0x7FFFu + ((b.u >> 16) & 1u);
  return (ul >> 16) | (uh & 0xFFFF0000u);
}

static __device__ inline float sigm(float x) { return 1.0f / (1.0f + expf(-x)); }

// ---------------------------------------------------------------------------
// LDS-staged fp32-in/out GEMM via bf16 WMMA:  C[M,N] = A[M,K] @ B[K,N] + bias
// Block: 128 threads (4 waves), 64x64 output tile, K-chunks of 32.
// LDS holds packed bf16 K-pairs: As2 row-major along K, Bt2 transposed [N][K].
// Wave w computes rows [16w,16w+16) x 64 cols = 4 WMMA tiles sharing one
// A fragment. CDNA5 wave32 fragment layouts (as packed dwords):
//   A frag dword v<4 -> K-pair 4h+v ; v>=4 -> K-pair 8+4h+(v-4)
//   B frag dword v<4 -> K-pair 4h+v ; v>=4 -> K-pair 8+4h+(v-4) at column 16j+lm
//   C/D (v8f): row = v + 8h, col = lane&15
// Interior tiles + full K-chunks use a guard-free coalesced fill; guards run
// only in the tail K-chunk / edge block columns (block-uniform branch).
// ---------------------------------------------------------------------------
#define BM 64
#define BN 64
#define BK 32

__global__ void __launch_bounds__(128)
wmma_gemm(const float* __restrict__ A, int lda,
          const float* __restrict__ B, int ldb,
          const float* __restrict__ bias,
          float* __restrict__ C, int ldc,
          int M, int N, int K)
{
  __shared__ unsigned As2[BM][BK / 2 + 1];  // packed bf16 pairs, +1 dword pad
  __shared__ unsigned Bt2[BN][BK / 2 + 1];  // B transposed, packed pairs

  const int tid  = threadIdx.x;       // 0..127
  const int wave = tid >> 5;          // 0..3
  const int l    = tid & 31;
  const int h    = l >> 4;            // lane half
  const int lm   = l & 15;
  const int bm0  = blockIdx.y * BM;
  const int bn0  = blockIdx.x * BN;

  const bool interior = (bm0 + BM <= M) && (bn0 + BN <= N);

  union F16x16 { v16bf v; unsigned u[8]; };

  v8f acc[4];
#pragma unroll
  for (int j = 0; j < 4; ++j) acc[j] = (v8f){0.f,0.f,0.f,0.f,0.f,0.f,0.f,0.f};

  for (int k0 = 0; k0 < K; k0 += BK) {
    if (interior && (k0 + BK <= K)) {
      // ---- FAST fill: no guards, fully coalesced ----
#pragma unroll
      for (int i = 0; i < 8; ++i) {            // A: 64 rows x 16 pairs
        int idx = tid + i * 128;
        int r = idx >> 4, p = idx & 15;
        const float* ap = A + (size_t)(bm0 + r) * lda + k0 + 2 * p;
        As2[r][p] = pack2(ap[0], ap[1]);
      }
#pragma unroll
      for (int i = 0; i < 8; ++i) {            // B: 16 pairs x 64 cols
        int idx = tid + i * 128;
        int p = idx >> 6, n = idx & 63;
        const float* bp = B + (size_t)(k0 + 2 * p) * ldb + bn0 + n;
        Bt2[n][p] = pack2(bp[0], bp[ldb]);
      }
    } else {
      // ---- EDGE fill: bounds-guarded (tail K-chunk / edge blocks only) ----
#pragma unroll
      for (int i = 0; i < 8; ++i) {
        int idx = tid + i * 128;
        int r = idx >> 4, p = idx & 15;
        int gr = bm0 + r, gc = k0 + 2 * p;
        float lo = (gr < M && gc     < K) ? A[(size_t)gr * lda + gc]     : 0.0f;
        float hi = (gr < M && gc + 1 < K) ? A[(size_t)gr * lda + gc + 1] : 0.0f;
        As2[r][p] = pack2(lo, hi);
      }
#pragma unroll
      for (int i = 0; i < 8; ++i) {
        int idx = tid + i * 128;
        int p = idx >> 6, n = idx & 63;
        int gr = k0 + 2 * p, gc = bn0 + n;
        float lo = (gr     < K && gc < N) ? B[(size_t)gr       * ldb + gc] : 0.0f;
        float hi = (gr + 1 < K && gc < N) ? B[(size_t)(gr + 1) * ldb + gc] : 0.0f;
        Bt2[n][p] = pack2(lo, hi);
      }
    }
    __syncthreads();

    // ---- A fragment: 8 LDS dword loads, no VALU ----
    F16x16 af;
    {
      const unsigned* ar = &As2[wave * 16 + lm][0];
#pragma unroll
      for (int v = 0; v < 4; ++v) {
        af.u[v]     = ar[4 * h + v];
        af.u[4 + v] = ar[8 + 4 * h + v];
      }
    }
    // ---- 4 WMMAs across the 64-wide N strip ----
#pragma unroll
    for (int j = 0; j < 4; ++j) {
      F16x16 bf_;
      const unsigned* br = &Bt2[j * 16 + lm][0];
#pragma unroll
      for (int v = 0; v < 4; ++v) {
        bf_.u[v]     = br[4 * h + v];
        bf_.u[4 + v] = br[8 + 4 * h + v];
      }
      acc[j] = __builtin_amdgcn_wmma_f32_16x16x32_bf16(
          false, af.v, false, bf_.v, (short)0, acc[j], false, false);
    }
    __syncthreads();
  }

#pragma unroll
  for (int j = 0; j < 4; ++j) {
#pragma unroll
    for (int v = 0; v < 8; ++v) {
      int row = bm0 + wave * 16 + v + 8 * h;
      int col = bn0 + j * 16 + lm;
      if (row < M && col < N)
        C[(size_t)row * ldc + col] = acc[j][v] + (bias ? bias[col] : 0.0f);
    }
  }
}

// ---------------------------------------------------------------------------
// Embedding gather: out[t, c] = emb[idx[t], c]
// ---------------------------------------------------------------------------
__global__ void gather_emb(const float* __restrict__ emb,
                           const int* __restrict__ idx,
                           float* __restrict__ out, int L)
{
  int i = blockIdx.x * blockDim.x + threadIdx.x;
  if (i >= L * H_) return;
  int t = i / H_, c = i - t * H_;
  out[i] = emb[(size_t)idx[t] * H_ + c];
}

// ---------------------------------------------------------------------------
// Persistent GRU scan (h kept in LDS). gi = precomputed x@Wih + bih, [L,900].
// ---------------------------------------------------------------------------
static __device__ void gru_scan_body(const float* __restrict__ gi,
                                     const float* __restrict__ Whh,
                                     const float* __restrict__ bhh,
                                     const float* __restrict__ hinit,
                                     float* __restrict__ out,
                                     int L, int ostride, int ooff)
{
  __shared__ float h[H_], h2[H_], gh[3 * H_];
  const int tid = threadIdx.x;
  if (tid < H_) h[tid] = hinit ? hinit[tid] : 0.0f;
  __syncthreads();
  for (int t = 0; t < L; ++t) {
    if (tid < 3 * H_) {
      float s = bhh[tid];
      for (int i = 0; i < H_; ++i) s += h[i] * Whh[i * (3 * H_) + tid];
      gh[tid] = s;
    }
    __syncthreads();
    if (tid < H_) {
      const float* g = gi + (size_t)t * (3 * H_);
      float r = sigm(g[tid]          + gh[tid]);
      float z = sigm(g[H_ + tid]     + gh[H_ + tid]);
      float n = tanhf(g[2 * H_ + tid] + r * gh[2 * H_ + tid]);
      float hn = (1.0f - z) * n + z * h[tid];
      h2[tid] = hn;
      out[(size_t)t * ostride + ooff + tid] = hn;
    }
    __syncthreads();
    if (tid < H_) h[tid] = h2[tid];
    __syncthreads();
  }
}

__global__ void __launch_bounds__(1024)
enc_scan(const float* gikf, const float* gikb, const float* giqf, const float* giqb,
         const float* WhhF, const float* WhhB, const float* bhhF, const float* bhhB,
         float* kb, float* cw)
{
  switch (blockIdx.x) {
    case 0: gru_scan_body(gikf, WhhF, bhhF, nullptr, kb, LKB, D_, 0);   break;
    case 1: gru_scan_body(gikb, WhhB, bhhB, nullptr, kb, LKB, D_, H_);  break;
    case 2: gru_scan_body(giqf, WhhF, bhhF, nullptr, cw, LQ,  D_, 0);   break;
    default:gru_scan_body(giqb, WhhB, bhhB, nullptr, cw, LQ,  D_, H_);  break;
  }
}

__global__ void __launch_bounds__(1024)
dec_scan(const float* gid, const float* Whh, const float* bhh,
         const float* h0, float* hs)
{
  gru_scan_body(gid, Whh, bhh, h0, hs, LT, H_, 0);
}

// ---------------------------------------------------------------------------
// q = [cw[0]@q_W + q_b , cw[-1]@q_W + q_b];  ms[0]=m0, cs[0]=c0
// ---------------------------------------------------------------------------
__global__ void __launch_bounds__(1024)
prep_q(const float* __restrict__ cw, const float* __restrict__ qW,
       const float* __restrict__ qb, float* __restrict__ qv,
       const float* __restrict__ m0, const float* __restrict__ c0,
       float* __restrict__ ms, float* __restrict__ cs)
{
  int tid = threadIdx.x;
  if (tid < H_) {
    float s = qb[tid];
    for (int k = 0; k < D_; ++k) s += cw[k] * qW[k * H_ + tid];
    qv[tid] = s;
  } else if (tid < D_) {
    int j = tid - H_;
    float s = qb[j];
    const float* cl = cw + (size_t)(LQ - 1) * D_;
    for (int k = 0; k < D_; ++k) s += cl[k] * qW[k * H_ + j];
    qv[tid] = s;
  }
  for (int j = tid; j < D_; j += 1024) { ms[j] = m0[j]; cs[j] = c0[j]; }
}

// ---------------------------------------------------------------------------
// MAC control unit for iteration `it` -> writes cs[it+1]
// ---------------------------------------------------------------------------
__global__ void __launch_bounds__(1024)
mac_control(const float* __restrict__ qv, float* __restrict__ cs,
            const float* __restrict__ cw,
            const float* __restrict__ cWdd, const float* __restrict__ cbdi,
            const float* __restrict__ cW2dd, const float* __restrict__ cbd,
            const float* __restrict__ cWd1, const float* __restrict__ cb1,
            int it)
{
  __shared__ float qi[D_], vv[D_], sm[LQ], red[1024];
  const int tid = threadIdx.x;
  const float* last_c = cs + it * D_;
  const float* Wi = cWdd + (size_t)it * D_ * D_;

  if (tid < D_) {
    float s = cbdi[it * D_ + tid];
    for (int k = 0; k < D_; ++k) s += qv[k] * Wi[k * D_ + tid];
    qi[tid] = s;
  }
  __syncthreads();
  if (tid < D_) {
    float s = cbd[tid];
    for (int k = 0; k < D_; ++k) s += qi[k] * cW2dd[k * D_ + tid];
    for (int k = 0; k < D_; ++k) s += last_c[k] * cW2dd[(D_ + k) * D_ + tid];
    vv[tid] = s * cWd1[tid];                 // cq[n]*cWd1[n]
  }
  __syncthreads();
  if (tid < LQ) {
    float s = cb1[0];
    for (int n = 0; n < D_; ++n) s += vv[n] * cw[(size_t)tid * D_ + n];
    sm[tid] = s;
  }
  __syncthreads();
  // softmax over LQ
  red[tid] = (tid < LQ) ? sm[tid] : -3.4e38f;
  __syncthreads();
  for (int s = 512; s > 0; s >>= 1) {
    if (tid < s) red[tid] = fmaxf(red[tid], red[tid + s]);
    __syncthreads();
  }
  float mx = red[0];
  __syncthreads();
  float e = (tid < LQ) ? expf(sm[tid] - mx) : 0.0f;
  red[tid] = e;
  __syncthreads();
  for (int s = 512; s > 0; s >>= 1) {
    if (tid < s) red[tid] += red[tid + s];
    __syncthreads();
  }
  float inv = 1.0f / red[0];
  __syncthreads();
  if (tid < LQ) sm[tid] = e * inv;
  __syncthreads();
  if (tid < D_) {
    float s = 0.f;
    for (int t = 0; t < LQ; ++t) s += sm[t] * cw[(size_t)t * D_ + tid];
    cs[(it + 1) * D_ + tid] = s / (float)LQ;
  }
}

// ---------------------------------------------------------------------------
// Read unit prep: w4c = c_i .* rW4 ; u = (last_m@rW1+rb1) .* (rW3_top @ w4c)
// ---------------------------------------------------------------------------
__global__ void __launch_bounds__(1024)
mac_read_prep(const float* __restrict__ ms, const float* __restrict__ cs, int it,
              const float* __restrict__ rW1, const float* __restrict__ rb1,
              const float* __restrict__ rW3, const float* __restrict__ rW4,
              float* __restrict__ u, float* __restrict__ w4c)
{
  __shared__ float w4s[D_];
  const int tid = threadIdx.x;
  const float* last_m = ms + it * D_;
  const float* c_i = cs + (it + 1) * D_;
  if (tid < D_) {
    float w = c_i[tid] * rW4[tid];
    w4s[tid] = w;
    w4c[tid] = w;
  }
  __syncthreads();
  if (tid < D_) {
    float mm = rb1[tid];
    for (int j = 0; j < D_; ++j) mm += last_m[j] * rW1[j * D_ + tid];
    float g = 0.f;
    for (int n = 0; n < D_; ++n) g += rW3[tid * D_ + n] * w4s[n];
    u[tid] = mm * g;
  }
}

// logits[t] = kb_r[t].u + Ibase[t].w4c + rb4
__global__ void mac_read_att(const float* __restrict__ kbr,
                             const float* __restrict__ ib,
                             const float* __restrict__ u,
                             const float* __restrict__ w4c,
                             const float* __restrict__ rb4,
                             float* __restrict__ logits)
{
  int t = blockIdx.x * blockDim.x + threadIdx.x;
  if (t >= LKB) return;
  float s = rb4[0];
  const float* kr = kbr + (size_t)t * D_;
  const float* ir = ib + (size_t)t * D_;
  for (int k = 0; k < D_; ++k) s += kr[k] * u[k] + ir[k] * w4c[k];
  logits[t] = s;
}

// softmax over LKB then m_new = (1/LKB) sum_t p[t]*kb[t,:]
__global__ void __launch_bounds__(1024)
mac_read_fin(float* __restrict__ logits, const float* __restrict__ kb,
             float* __restrict__ m_new)
{
  __shared__ float red[1024];
  const int tid = threadIdx.x;
  float m = -3.4e38f;
  for (int t = tid; t < LKB; t += 1024) m = fmaxf(m, logits[t]);
  red[tid] = m;
  __syncthreads();
  for (int s = 512; s > 0; s >>= 1) {
    if (tid < s) red[tid] = fmaxf(red[tid], red[tid + s]);
    __syncthreads();
  }
  float mx = red[0];
  __syncthreads();
  float sum = 0.f;
  for (int t = tid; t < LKB; t += 1024) {
    float e = expf(logits[t] - mx);
    logits[t] = e;
    sum += e;
  }
  red[tid] = sum;
  __syncthreads();
  for (int s = 512; s > 0; s >>= 1) {
    if (tid < s) red[tid] += red[tid + s];
    __syncthreads();
  }
  float inv = 1.0f / red[0];
  __syncthreads();
  for (int t = tid; t < LKB; t += 1024) logits[t] *= inv;
  __syncthreads();
  if (tid < D_) {
    float s = 0.f;
    for (int t = 0; t < LKB; ++t) s += logits[t] * kb[(size_t)t * D_ + tid];
    m_new[tid] = s / (float)LKB;
  }
}

// ---------------------------------------------------------------------------
// MAC write unit -> ms[it+1]
// ---------------------------------------------------------------------------
__global__ void __launch_bounds__(1024)
mac_write(const float* __restrict__ m_new, float* __restrict__ ms,
          const float* __restrict__ cs, int it,
          const float* __restrict__ wW1, const float* __restrict__ wb1,
          const float* __restrict__ wW2, const float* __restrict__ wb2,
          const float* __restrict__ wW3, const float* __restrict__ wb3,
          const float* __restrict__ wW4, const float* __restrict__ wb4)
{
  __shared__ float mm[D_], msa[D_], aw[8];
  const int tid = threadIdx.x;
  const float* last_m = ms + it * D_;
  const float* c_i = cs + (it + 1) * D_;
  const int kcnt = it + 1;

  if (tid < D_) {
    float s = wb1[tid];
    for (int k = 0; k < D_; ++k) s += m_new[k] * wW1[k * D_ + tid];
    for (int k = 0; k < D_; ++k) s += last_m[k] * wW1[(D_ + k) * D_ + tid];
    mm[tid] = s;
  }
  if (tid >= D_ && tid < D_ + kcnt) {
    int r = tid - D_;
    float s = wb2[0];
    const float* cr = cs + r * D_;
    for (int n = 0; n < D_; ++n) s += cr[n] * c_i[n] * wW2[n];
    aw[r] = s;
  }
  __syncthreads();
  if (tid == 0) {
    float mx = -3.4e38f;
    for (int r = 0; r < kcnt; ++r) mx = fmaxf(mx, aw[r]);
    float sum = 0.f;
    for (int r = 0; r < kcnt; ++r) { aw[r] = expf(aw[r] - mx); sum += aw[r]; }
    for (int r = 0; r < kcnt; ++r) aw[r] /= sum;
  }
  __syncthreads();
  if (tid < D_) {
    float s = 0.f;
    for (int r = 0; r < kcnt; ++r) s += aw[r] * ms[r * D_ + tid];
    msa[tid] = s / (float)kcnt;
  }
  __syncthreads();
  if (tid < D_) {
    float m2 = wb3[tid];
    for (int k = 0; k < D_; ++k) m2 += msa[k] * wW3[k * D_ + tid];
    for (int k = 0; k < D_; ++k) m2 += mm[k] * wW3[(D_ + k) * D_ + tid];
    float g = wb4[tid];
    for (int k = 0; k < D_; ++k) g += c_i[k] * wW4[k * D_ + tid];
    g = sigm(g);
    ms[(it + 1) * D_ + tid] = g * last_m[tid] + (1.0f - g) * m2;
  }
}

// ---------------------------------------------------------------------------
// reasoning = tanh([q, m7]); mid = [kb[0], cw[0], reasoning]; h0 = mid@s_W+s_b
// ---------------------------------------------------------------------------
__global__ void __launch_bounds__(1024)
reason_h0(const float* __restrict__ qv, const float* __restrict__ ms,
          const float* __restrict__ kb, const float* __restrict__ cw,
          const float* __restrict__ sW, const float* __restrict__ sb,
          float* __restrict__ h0)
{
  __shared__ float mid[4 * D_];
  const int tid = threadIdx.x;
  const float* m7 = ms + NMAC * D_;
  for (int j = tid; j < D_; j += 1024) { mid[j] = kb[j]; mid[D_ + j] = cw[j]; }
  for (int j = tid; j < 2 * D_; j += 1024) {
    float x = (j < D_) ? qv[j] : m7[j - D_];
    mid[2 * D_ + j] = tanhf(x);
  }
  __syncthreads();
  if (tid < H_) {
    float s = sb[tid];
    for (int k = 0; k < 4 * D_; ++k) s += mid[k] * sW[k * H_ + tid];
    h0[tid] = s;
  }
}

// ---------------------------------------------------------------------------
static inline int ceilDiv(int a, int b) { return (a + b - 1) / b; }

extern "C" void kernel_launch(void* const* d_in, const int* in_sizes, int n_in,
                              void* d_out, int out_size, void* d_ws, size_t ws_size,
                              hipStream_t stream)
{
  (void)in_sizes; (void)n_in; (void)out_size; (void)ws_size;
  const int*   kb_in  = (const int*)d_in[0];
  const int*   q_in   = (const int*)d_in[1];
  const int*   t_in   = (const int*)d_in[2];
  const float* emb    = (const float*)d_in[3];
  const float* efWih  = (const float*)d_in[4];
  const float* efWhh  = (const float*)d_in[5];
  const float* efbih  = (const float*)d_in[6];
  const float* efbhh  = (const float*)d_in[7];
  const float* ebWih  = (const float*)d_in[8];
  const float* ebWhh  = (const float*)d_in[9];
  const float* ebbih  = (const float*)d_in[10];
  const float* ebbhh  = (const float*)d_in[11];
  const float* dWih   = (const float*)d_in[12];
  const float* dWhh   = (const float*)d_in[13];
  const float* dbih   = (const float*)d_in[14];
  const float* dbhh   = (const float*)d_in[15];
  const float* qW     = (const float*)d_in[16];
  const float* qb     = (const float*)d_in[17];
  const float* sW     = (const float*)d_in[18];
  const float* sb     = (const float*)d_in[19];
  const float* oW     = (const float*)d_in[20];
  const float* ob     = (const float*)d_in[21];
  const float* cWdd   = (const float*)d_in[22];
  const float* cbdi   = (const float*)d_in[23];
  const float* cW2dd  = (const float*)d_in[24];
  const float* cbd    = (const float*)d_in[25];
  const float* cWd1   = (const float*)d_in[26];
  const float* cb1    = (const float*)d_in[27];
  const float* rW1    = (const float*)d_in[28];
  const float* rb1    = (const float*)d_in[29];
  const float* rW2    = (const float*)d_in[30];
  const float* rb2    = (const float*)d_in[31];
  const float* rW3    = (const float*)d_in[32];
  const float* rb3    = (const float*)d_in[33];
  const float* rW4    = (const float*)d_in[34];
  const float* rb4    = (const float*)d_in[35];
  const float* wW1    = (const float*)d_in[36];
  const float* wb1    = (const float*)d_in[37];
  const float* wW2    = (const float*)d_in[38];
  const float* wb2    = (const float*)d_in[39];
  const float* wW3    = (const float*)d_in[40];
  const float* wb3    = (const float*)d_in[41];
  const float* wW4    = (const float*)d_in[42];
  const float* wb4    = (const float*)d_in[43];
  const float* m0     = (const float*)d_in[44];
  const float* c0     = (const float*)d_in[45];

  float* ws = (float*)d_ws;
  size_t o = 0;
  float* KBE  = ws + o; o += (size_t)LKB * H_;
  float* QE   = ws + o; o += (size_t)LQ  * H_;
  float* TGE  = ws + o; o += (size_t)LT  * H_;
  float* GIKF = ws + o; o += (size_t)LKB * 3 * H_;
  float* GIKB = ws + o; o += (size_t)LKB * 3 * H_;
  float* GIQF = ws + o; o += (size_t)LQ  * 3 * H_;
  float* GIQB = ws + o; o += (size_t)LQ  * 3 * H_;
  float* GID  = ws + o; o += (size_t)LT  * 3 * H_;
  float* KB   = ws + o; o += (size_t)LKB * D_;
  float* CW   = ws + o; o += (size_t)LQ  * D_;
  float* KBR  = ws + o; o += (size_t)LKB * D_;
  float* IB   = ws + o; o += (size_t)LKB * D_;
  float* QV   = ws + o; o += D_;
  float* MS   = ws + o; o += 8 * D_;
  float* CS   = ws + o; o += 8 * D_;
  float* UU   = ws + o; o += D_;
  float* W4C  = ws + o; o += D_;
  float* LG   = ws + o; o += LKB;
  float* MNEW = ws + o; o += D_;
  float* H0   = ws + o; o += H_;
  float* HS   = ws + o; o += (size_t)LT * H_;

  // 1) embedding gathers
  gather_emb<<<ceilDiv(LKB * H_, 256), 256, 0, stream>>>(emb, kb_in, KBE, LKB);
  gather_emb<<<ceilDiv(LQ  * H_, 256), 256, 0, stream>>>(emb, q_in,  QE,  LQ);
  gather_emb<<<ceilDiv(LT  * H_, 256), 256, 0, stream>>>(emb, t_in,  TGE, LT);

  // 2) batched GRU input-gate GEMMs (WMMA)
  {
    dim3 gk(ceilDiv(3 * H_, BN), ceilDiv(LKB, BM));
    wmma_gemm<<<gk, 128, 0, stream>>>(KBE, H_, efWih, 3 * H_, efbih, GIKF, 3 * H_, LKB, 3 * H_, H_);
    wmma_gemm<<<gk, 128, 0, stream>>>(KBE, H_, ebWih, 3 * H_, ebbih, GIKB, 3 * H_, LKB, 3 * H_, H_);
    dim3 gq(ceilDiv(3 * H_, BN), ceilDiv(LQ, BM));
    wmma_gemm<<<gq, 128, 0, stream>>>(QE, H_, efWih, 3 * H_, efbih, GIQF, 3 * H_, LQ, 3 * H_, H_);
    wmma_gemm<<<gq, 128, 0, stream>>>(QE, H_, ebWih, 3 * H_, ebbih, GIQB, 3 * H_, LQ, 3 * H_, H_);
    dim3 gd(ceilDiv(3 * H_, BN), ceilDiv(LT, BM));
    wmma_gemm<<<gd, 128, 0, stream>>>(TGE, H_, dWih, 3 * H_, dbih, GID, 3 * H_, LT, 3 * H_, H_);
  }

  // 3) four independent encoder scans in one launch
  enc_scan<<<4, 1024, 0, stream>>>(GIKF, GIKB, GIQF, GIQB,
                                   efWhh, ebWhh, efbhh, ebbhh, KB, CW);

  // 4) loop-invariant MAC read-unit GEMMs (WMMA)
  {
    dim3 g(ceilDiv(D_, BN), ceilDiv(LKB, BM));
    wmma_gemm<<<g, 128, 0, stream>>>(KB, D_, rW2, D_, rb2, KBR, D_, LKB, D_, D_);
    // Ibase = kb @ rW3[600:1200,:] + rb3
    wmma_gemm<<<g, 128, 0, stream>>>(KB, D_, rW3 + (size_t)D_ * D_, D_, rb3, IB, D_, LKB, D_, D_);
  }

  // 5) q vector + initial memory/control state
  prep_q<<<1, 1024, 0, stream>>>(CW, qW, qb, QV, m0, c0, MS, CS);

  // 6) MAC reasoning loop
  for (int it = 0; it < NMAC; ++it) {
    mac_control<<<1, 1024, 0, stream>>>(QV, CS, CW, cWdd, cbdi, cW2dd, cbd, cWd1, cb1, it);
    mac_read_prep<<<1, 1024, 0, stream>>>(MS, CS, it, rW1, rb1, rW3, rW4, UU, W4C);
    mac_read_att<<<LKB / 128, 128, 0, stream>>>(KBR, IB, UU, W4C, rb4, LG);
    mac_read_fin<<<1, 1024, 0, stream>>>(LG, KB, MNEW);
    mac_write<<<1, 1024, 0, stream>>>(MNEW, MS, CS, it,
                                      wW1, wb1, wW2, wb2, wW3, wb3, wW4, wb4);
  }

  // 7) reasoning -> decoder initial hidden
  reason_h0<<<1, 1024, 0, stream>>>(QV, MS, KB, CW, sW, sb, H0);

  // 8) decoder scan
  dec_scan<<<1, 1024, 0, stream>>>(GID, dWhh, dbhh, H0, HS);

  // 9) output projection hs @ o_W + o_b -> d_out   (dominant GEMM, WMMA)
  {
    dim3 g(ceilDiv(VOC, BN), ceilDiv(LT, BM));
    wmma_gemm<<<g, 128, 0, stream>>>(HS, H_, oW, VOC, ob, (float*)d_out, VOC, LT, VOC, H_);
  }
}
